// SharedSpecialistMoEFFN_72103910965742
// MI455X (gfx1250) — compile-verified, matrix operand order checked by reading, not run on hardware
//
#include <hip/hip_runtime.h>
#include <math.h>

// ---------------------------------------------------------------------------
// SharedSpecialistMoEFFN for gfx1250 (MI455X), bf16 WMMA path.
//   out = gelu(x@W1s+b1s)@W2s+b2s  +  sum_e combine[t,e]*(gelu(x@W1e+b1e)@W2e+b2e)
// combine = renormalized top-2 of softmax(x@Wr+br); dense (zeros elsewhere).
// GEMMs: 128x128 block tile, K-step 32, 8 wave32s (4Mx2N), each wave 32x64 via
// v_wmma_f32_16x16x32_bf16. A tiles: async global->LDS (ASYNCcnt). B tiles:
// fp32 weights converted to bf16 in flight. Double-buffered LDS.
// ---------------------------------------------------------------------------

#define D_MODEL 1024
#define D_FF    4096
#define N_EXP   8
#define T_TOTAL 8192
#define CHUNK_M 2048

typedef __bf16 bf16;
typedef __bf16 v16bf  __attribute__((ext_vector_type(16)));
typedef __bf16 bf16x8 __attribute__((ext_vector_type(8)));
typedef float  v8f    __attribute__((ext_vector_type(8)));

#if __has_builtin(__builtin_amdgcn_global_load_async_to_lds_b128)
#define ASYNC_A 1
// Exact param types per hipcc diagnostic: v4i in AS1 (src), LDS dst in AS3.
typedef int v4i __attribute__((vector_size(4 * sizeof(int))));
typedef __attribute__((address_space(1))) v4i* as1_v4i_ptr;
typedef __attribute__((address_space(3))) v4i* as3_v4i_ptr;
#else
#define ASYNC_A 0
#endif

__device__ __forceinline__ void async_wait0() {
#if ASYNC_A
#if __has_builtin(__builtin_amdgcn_s_wait_asynccnt)
    __builtin_amdgcn_s_wait_asynccnt(0);
#else
    asm volatile("s_wait_asynccnt 0x0" ::: "memory");
#endif
#endif
}

// Branch-free erf (Abramowitz-Stegun 7.1.26) -> exact-style GELU, one v_exp.
__device__ __forceinline__ float gelu_exact(float x) {
    float z  = x * 0.70710678118654752f;
    float az = fabsf(z);
    float t  = 1.0f / (1.0f + 0.3275911f * az);
    float y  = t * (0.254829592f +
               t * (-0.284496736f +
               t * (1.421413741f +
               t * (-1.453152027f +
               t * 1.061405429f))));
    float er = copysignf(1.0f - y * __expf(-az * az), z);
    return 0.5f * x * (1.0f + er);
}

// ---------------------------------------------------------------------------
// fp32 -> bf16 conversion (for x)
// ---------------------------------------------------------------------------
__global__ void cvt_f32_to_bf16_kernel(const float* __restrict__ in,
                                       bf16* __restrict__ out, int n) {
    int i = blockIdx.x * blockDim.x + threadIdx.x;
    int stride = gridDim.x * blockDim.x;
    for (; i < n; i += stride) out[i] = (bf16)in[i];
}

// ---------------------------------------------------------------------------
// Router: one wave32 per token. logits = x@Wr + br; softmax; top-2; renorm;
// scatter into dense combine[T, N_EXP].
// ---------------------------------------------------------------------------
__global__ __launch_bounds__(256)
void router_kernel(const float* __restrict__ x, const float* __restrict__ rw,
                   const float* __restrict__ rb, float* __restrict__ combine,
                   int T) {
    const int lane = threadIdx.x & 31;
    const int wave = threadIdx.x >> 5;
    const int t = blockIdx.x * 8 + wave;
    if (t >= T) return;

    float acc[N_EXP];
#pragma unroll
    for (int e = 0; e < N_EXP; ++e) acc[e] = 0.0f;

    const float* xr = x + (size_t)t * D_MODEL;
    for (int k = lane; k < D_MODEL; k += 32) {
        float xv = xr[k];
        const float* wr = rw + (size_t)k * N_EXP;
#pragma unroll
        for (int e = 0; e < N_EXP; ++e) acc[e] += xv * wr[e];
    }
#pragma unroll
    for (int off = 16; off >= 1; off >>= 1) {
#pragma unroll
        for (int e = 0; e < N_EXP; ++e)
            acc[e] += __shfl_down(acc[e], off, 32);
    }
    if (lane == 0) {
        float p[N_EXP];
        float mx = -1e30f;
#pragma unroll
        for (int e = 0; e < N_EXP; ++e) { p[e] = acc[e] + rb[e]; mx = fmaxf(mx, p[e]); }
        float s = 0.0f;
#pragma unroll
        for (int e = 0; e < N_EXP; ++e) { p[e] = __expf(p[e] - mx); s += p[e]; }
        float inv = 1.0f / s;
#pragma unroll
        for (int e = 0; e < N_EXP; ++e) p[e] *= inv;

        int i1 = 0;
#pragma unroll
        for (int e = 1; e < N_EXP; ++e) if (p[e] > p[i1]) i1 = e;
        int i2 = (i1 == 0) ? 1 : 0;
#pragma unroll
        for (int e = 0; e < N_EXP; ++e) if (e != i1 && p[e] > p[i2]) i2 = e;

        float norm = 1.0f / (p[i1] + p[i2] + 1e-9f);
        float* crow = combine + (size_t)t * N_EXP;
#pragma unroll
        for (int e = 0; e < N_EXP; ++e) crow[e] = 0.0f;
        crow[i1] = p[i1] * norm;
        crow[i2] = p[i2] * norm;
    }
}

// ---------------------------------------------------------------------------
// Tiled bf16 WMMA GEMM with fused epilogue (double-buffered LDS).
//   A: [M,K] bf16 (row-major), W: [K,N] fp32 (row-major, converted in-flight).
// MODE 0: Hout[m,n]  = bf16(gelu(acc + bias[n]))
// MODE 1: Fout[m,n]  = acc + bias[n]
// MODE 2: Fout[m,n] += cw[m*8+expert] * (acc + bias[n])
// ---------------------------------------------------------------------------
template <int MODE>
__global__ __launch_bounds__(256)
void ffn_gemm_kernel(const bf16* __restrict__ A, const float* __restrict__ W,
                     const float* __restrict__ bias, bf16* __restrict__ Hout,
                     float* __restrict__ Fout, const float* __restrict__ cw,
                     int expert, int M, int N, int K) {
    constexpr int LDSS = 40;  // bf16 elems per row (80B: 16B-aligned, padded)
    __shared__ bf16 ldsA[2][128 * LDSS];
    __shared__ bf16 ldsB[2][128 * LDSS];

    const int tid  = threadIdx.x;
    const int lane = tid & 31;
    const int wave = tid >> 5;
    const int wm   = wave >> 1;            // 0..3 : wave M index
    const int wn   = wave & 1;             // 0..1 : wave N index
    const int frow = lane & 15;            // fragment row (A) / col (B)
    const int kb   = (lane >> 4) * 8;      // K-base for this half-wave

    const int blockM = blockIdx.y * 128;
    const int blockN = blockIdx.x * 128;

    // Stage tile kt into LDS buffer `buf`.
    auto stage = [&](int buf, int kt) {
        bf16* lA = &ldsA[buf][0];
        bf16* lB = &ldsB[buf][0];
        // ---- A tile [128 x 32] bf16: 512 16B chunks, 2 per thread ----
#pragma unroll
        for (int it = 0; it < 2; ++it) {
            int c = tid + it * 256;
            int m = c >> 2, cc = c & 3;
            const bf16x8* src = (const bf16x8*)(A + (size_t)(blockM + m) * K +
                                                (size_t)kt * 32 + cc * 8);
#if ASYNC_A
            __builtin_amdgcn_global_load_async_to_lds_b128(
                (as1_v4i_ptr)src, (as3_v4i_ptr)&lA[m * LDSS + cc * 8], 0, 0);
#else
            *(bf16x8*)(&lA[m * LDSS + cc * 8]) = *src;
#endif
        }
        // ---- B tile [32 x 128] f32 -> bf16, transposed to [N][K] ----
#pragma unroll
        for (int it = 0; it < 4; ++it) {
            int c = tid + it * 256;
            int k = c >> 5, ncc = c & 31;
            const float4 v = *(const float4*)(W + (size_t)(kt * 32 + k) * N +
                                              blockN + ncc * 4);
            int nb = ncc * 4;
            lB[(nb + 0) * LDSS + k] = (bf16)v.x;
            lB[(nb + 1) * LDSS + k] = (bf16)v.y;
            lB[(nb + 2) * LDSS + k] = (bf16)v.z;
            lB[(nb + 3) * LDSS + k] = (bf16)v.w;
        }
    };

    v8f acc[2][4];
#pragma unroll
    for (int i = 0; i < 2; ++i)
#pragma unroll
        for (int j = 0; j < 4; ++j)
#pragma unroll
            for (int r = 0; r < 8; ++r) acc[i][j][r] = 0.0f;

    const int ksteps = K >> 5;
    stage(0, 0);
    for (int kt = 0; kt < ksteps; ++kt) {
        const int buf = kt & 1;
        async_wait0();       // A-tile async loads (this wave) have landed
        __syncthreads();     // (compiler adds dscnt wait for B-tile ds_stores)
        if (kt + 1 < ksteps) stage(buf ^ 1, kt + 1);

        // ---- fragments: ISA 16-bit 16x32 layout, elem i -> K = kb+i (+8 hi) ----
        v16bf afrag[2], bfrag[4];
#pragma unroll
        for (int i = 0; i < 2; ++i) {
            int base = (wm * 32 + i * 16 + frow) * LDSS + kb;
            bf16x8 lo = *(const bf16x8*)(&ldsA[buf][base]);
            bf16x8 hi = *(const bf16x8*)(&ldsA[buf][base + 16]);
#pragma unroll
            for (int e = 0; e < 8; ++e) { afrag[i][e] = lo[e]; afrag[i][e + 8] = hi[e]; }
        }
#pragma unroll
        for (int j = 0; j < 4; ++j) {
            int base = (wn * 64 + j * 16 + frow) * LDSS + kb;
            bf16x8 lo = *(const bf16x8*)(&ldsB[buf][base]);
            bf16x8 hi = *(const bf16x8*)(&ldsB[buf][base + 16]);
#pragma unroll
            for (int e = 0; e < 8; ++e) { bfrag[j][e] = lo[e]; bfrag[j][e + 8] = hi[e]; }
        }
#pragma unroll
        for (int i = 0; i < 2; ++i)
#pragma unroll
            for (int j = 0; j < 4; ++j)
                acc[i][j] = __builtin_amdgcn_wmma_f32_16x16x32_bf16(
                    false, afrag[i], false, bfrag[j], (short)0, acc[i][j],
                    false, false);
        __syncthreads();
    }

    // ---- epilogue: C/D f32 layout — VGPR r: row r (+8 for upper half-wave) ----
    const int mbase  = blockM + wm * 32;
    const int nbase  = blockN + wn * 64;
    const int rowadd = 8 * (lane >> 4);
    const int coladd = lane & 15;

    float bias4[4];
#pragma unroll
    for (int j = 0; j < 4; ++j) bias4[j] = bias[nbase + j * 16 + coladd];

#pragma unroll
    for (int i = 0; i < 2; ++i) {
#pragma unroll
        for (int r = 0; r < 8; ++r) {
            const int row = mbase + i * 16 + r + rowadd;
            const size_t rbase = (size_t)row * N;
            float wexp = 0.0f;
            if (MODE == 2) wexp = cw[(size_t)row * N_EXP + expert];
#pragma unroll
            for (int j = 0; j < 4; ++j) {
                const int col = nbase + j * 16 + coladd;
                float v = acc[i][j][r] + bias4[j];
                if (MODE == 0) {
                    Hout[rbase + col] = (bf16)gelu_exact(v);
                } else if (MODE == 1) {
                    Fout[rbase + col] = v;
                } else {
                    Fout[rbase + col] += wexp * v;
                }
            }
        }
    }
}

// ---------------------------------------------------------------------------
// Host-side orchestration
// ---------------------------------------------------------------------------
extern "C" void kernel_launch(void* const* d_in, const int* in_sizes, int n_in,
                              void* d_out, int out_size, void* d_ws, size_t ws_size,
                              hipStream_t stream) {
    const float* x         = (const float*)d_in[0];   // [T, 1024]
    const float* shared_w1 = (const float*)d_in[1];   // [1024, 4096]
    const float* shared_b1 = (const float*)d_in[2];   // [4096]
    const float* shared_w2 = (const float*)d_in[3];   // [4096, 1024]
    const float* shared_b2 = (const float*)d_in[4];   // [1024]
    const float* router_w  = (const float*)d_in[5];   // [1024, 8]
    const float* router_b  = (const float*)d_in[6];   // [8]
    const float* exp_w1    = (const float*)d_in[7];   // [8, 1024, 4096]
    const float* exp_b1    = (const float*)d_in[8];   // [8, 4096]
    const float* exp_w2    = (const float*)d_in[9];   // [8, 4096, 1024]
    const float* exp_b2    = (const float*)d_in[10];  // [8, 1024]
    float* out = (float*)d_out;                       // [T, 1024]

    // Workspace layout (~32.3 MB):
    //   x_bf   : T * D_MODEL       bf16   (16 MB)
    //   combine: T * N_EXP         f32    (256 KB)
    //   h_bf   : CHUNK_M * D_FF    bf16   (16 MB, reused per GEMM pair)
    char* ws = (char*)d_ws;
    bf16*  x_bf    = (bf16*)ws;
    float* combine = (float*)(ws + (size_t)T_TOTAL * D_MODEL * sizeof(bf16));
    bf16*  h_bf    = (bf16*)((char*)combine + (size_t)T_TOTAL * N_EXP * sizeof(float));

    // 1) x -> bf16
    cvt_f32_to_bf16_kernel<<<4096, 256, 0, stream>>>(x, x_bf, T_TOTAL * D_MODEL);

    // 2) router -> dense combine weights
    router_kernel<<<T_TOTAL / 8, 256, 0, stream>>>(x, router_w, router_b,
                                                   combine, T_TOTAL);

    // 3) FFNs, token chunks of CHUNK_M
    const dim3 blk(256);
    const dim3 grid1(D_FF / 128, CHUNK_M / 128);     // GEMM1: N=4096
    const dim3 grid2(D_MODEL / 128, CHUNK_M / 128);  // GEMM2: N=1024

    for (int c = 0; c < T_TOTAL / CHUNK_M; ++c) {
        const bf16* xA    = x_bf + (size_t)c * CHUNK_M * D_MODEL;
        float*      outC  = out + (size_t)c * CHUNK_M * D_MODEL;
        const float* cwC  = combine + (size_t)c * CHUNK_M * N_EXP;

        // shared FFN (writes out)
        ffn_gemm_kernel<0><<<grid1, blk, 0, stream>>>(
            xA, shared_w1, shared_b1, h_bf, nullptr, nullptr, 0,
            CHUNK_M, D_FF, D_MODEL);
        ffn_gemm_kernel<1><<<grid2, blk, 0, stream>>>(
            h_bf, shared_w2, shared_b2, nullptr, outC, nullptr, 0,
            CHUNK_M, D_MODEL, D_FF);

        // experts (accumulate combine[t,e] * FFN_e into out)
        for (int e = 0; e < N_EXP; ++e) {
            const float* w1 = exp_w1 + (size_t)e * D_MODEL * D_FF;
            const float* b1 = exp_b1 + (size_t)e * D_FF;
            const float* w2 = exp_w2 + (size_t)e * D_FF * D_MODEL;
            const float* b2 = exp_b2 + (size_t)e * D_MODEL;
            ffn_gemm_kernel<0><<<grid1, blk, 0, stream>>>(
                xA, w1, b1, h_bf, nullptr, nullptr, 0,
                CHUNK_M, D_FF, D_MODEL);
            ffn_gemm_kernel<2><<<grid2, blk, 0, stream>>>(
                h_bf, w2, b2, nullptr, outC, cwC, e,
                CHUNK_M, D_MODEL, D_FF);
        }
    }
}
